// StructuredGenerator_18176301597366
// MI455X (gfx1250) — matrix operand order repaired and checked by reading, not compiled, not run on hardware
//
#include <hip/hip_runtime.h>
#include <hip/hip_bf16.h>
#include <math.h>

// ---------------------------------------------------------------------------
// StructuredGenerator: GRUCell(128) autoregressive -> 2-layer biLSTM(800)
//                      -> atan2 angle projection.  bf16 WMMA everywhere.
// B=128 T=200 IN=128 H=800, gates 4H=3200.
// ---------------------------------------------------------------------------

#define B_   128
#define T_   200
#define IN_  128
#define H_   800
#define G4_  3200   // 4*H

typedef __attribute__((ext_vector_type(16))) __bf16 v16bf;
typedef __attribute__((ext_vector_type(8)))  __bf16 v8bf;
typedef __attribute__((ext_vector_type(8)))  float  v8f;

#if defined(__HIP_DEVICE_COMPILE__) && \
    __has_builtin(__builtin_amdgcn_tensor_load_to_lds) && \
    __has_builtin(__builtin_amdgcn_s_wait_tensorcnt)
#define USE_TDM 1
#endif

__device__ __forceinline__ float sigf(float x) { return 1.0f / (1.0f + __expf(-x)); }

__device__ __forceinline__ v16bf mk16(v8bf lo, v8bf hi) {
    v16bf r;
#pragma unroll
    for (int i = 0; i < 8; i++) { r[i] = lo[i]; r[i + 8] = hi[i]; }
    return r;
}

// ---------------------------------------------------------------------------
// packing kernels
// ---------------------------------------------------------------------------
__global__ void cvt_f32_bf16(__bf16* __restrict__ dst, const float* __restrict__ src, int n) {
    int i = blockIdx.x * blockDim.x + threadIdx.x;
    if (i < n) dst[i] = (__bf16)src[i];
}

// Wcat[r][c] = c<din ? Wih[r][c] : Whh[r][c-din]  ; row-major [3200][din+800]
__global__ void pack_wcat(__bf16* __restrict__ dst, const float* __restrict__ wih,
                          const float* __restrict__ whh, int din) {
    const int Kt = din + H_;
    const long n = (long)G4_ * Kt;
    for (long i = (long)blockIdx.x * blockDim.x + threadIdx.x; i < n;
         i += (long)gridDim.x * blockDim.x) {
        int r = (int)(i / Kt), c = (int)(i % Kt);
        float v = (c < din) ? wih[(long)r * din + c] : whh[(long)r * H_ + (c - din)];
        dst[i] = (__bf16)v;
    }
}

__global__ void bias_sum(float* __restrict__ dst, const float* __restrict__ a,
                         const float* __restrict__ b, int n) {
    int i = blockIdx.x * blockDim.x + threadIdx.x;
    if (i < n) dst[i] = a[i] + b[i];
}

// ---------------------------------------------------------------------------
// TDM helper: load the 3-D weight tile [gates=4][rows=16][k=32] bf16 (4KB)
// into LDS.  dim0 = k (stride 1), dim1 = row (stride Kt), dim2 = gate
// (stride 800*Kt).  D# packed per CDNA5 ISA sec 8.3-8.6 (group0/1/2/3).
// This toolchain exposes the 6-arg builtin (clang-23 form):
//   (u32x4 g0, i32x8 g1, i32x4 g2, i32x4 g3, i32x8 extra, i32 cpol)
// ---------------------------------------------------------------------------
#ifdef USE_TDM
typedef unsigned int u32x4 __attribute__((ext_vector_type(4)));
typedef int          i32x8 __attribute__((ext_vector_type(8)));
typedef int          i32x4 __attribute__((ext_vector_type(4)));

__device__ __forceinline__ unsigned lds_off_of(void* p) {
    return (unsigned)(uintptr_t)(__attribute__((address_space(3))) void*)p;
}

__device__ __forceinline__ void tdm_load_B(const __bf16* W, int Kt, int n0, int k0,
                                           unsigned lds_byte_off) {
    unsigned long long ga =
        (unsigned long long)(uintptr_t)W + 2ull * ((unsigned long long)n0 * Kt + k0);
    unsigned long long s0 = (unsigned long long)Kt;          // dim0 stride (elems)
    unsigned long long s1 = 800ull * (unsigned long long)Kt; // dim1 stride (elems)
    u32x4 g0 = { 1u,                                   // count=1 user descriptor
                 lds_byte_off,                         // lds_addr [63:32]
                 (unsigned)(ga & 0xFFFFFFFFu),         // global_addr lo
                 (unsigned)((ga >> 32) & 0x01FFFFFFu) | (2u << 30) }; // ga hi | type=2
    i32x8 g1 = { (int)(1u << 16),                                   // data_size=2B
                 (int)(((unsigned)Kt & 0xFFFFu) << 16),             // tensor_dim0 lo
                 (int)((((unsigned)Kt >> 16) & 0xFFFFu) | (3200u << 16)), // d0 hi|d1 lo
                 (int)(32u << 16),                                  // d1 hi=0 | tile_dim0=32
                 (int)(16u | (4u << 16)),                           // tile_dim1 | tile_dim2
                 (int)(unsigned)(s0 & 0xFFFFFFFFu),                 // dim0_stride lo
                 (int)(((unsigned)(s0 >> 32) & 0xFFFFu) |
                       ((unsigned)(s1 & 0xFFFFu) << 16)),           // d0s hi | d1s lo
                 (int)(unsigned)(s1 >> 16) };                       // dim1_stride hi
    i32x4 g2 = { 4, 0, 0, 0 };  // tensor_dim2 = 4 gates
    i32x4 g3 = { 0, 0, 0, 0 };
    i32x8 gx = { 0, 0, 0, 0, 0, 0, 0, 0 };
    __builtin_amdgcn_tensor_load_to_lds(g0, g1, g2, g3, gx, 0);
}
#endif

// ---------------------------------------------------------------------------
// Autoregressive GRU: single workgroup, 200 steps.  All weights (192KB bf16)
// preloaded to LDS; h ping-pongs in LDS -> the whole recurrence is
// LDS-resident (only y0 stores hit global).  x_{t+1} = h_t per reference.
// ---------------------------------------------------------------------------
__global__ __launch_bounds__(256) void gru_kernel(
    const float* __restrict__ latent, const __bf16* __restrict__ Wih,
    const __bf16* __restrict__ Whh, const float* __restrict__ bih,
    const float* __restrict__ bhh, const int* __restrict__ lengths,
    __bf16* __restrict__ y0 /* [B][T][IN] */) {
    extern __shared__ __bf16 smem[];
    __bf16* bufA = smem;              // latent bf16, 128x128
    __bf16* h0   = smem + 16384;      // ping
    __bf16* h1   = smem + 32768;      // pong
    __bf16* sWih = smem + 49152;      // [384][128]
    __bf16* sWhh = smem + 98304;      // [384][128]

    const int tid = threadIdx.x, lane = tid & 31, wave = tid >> 5;
    for (int i = tid; i < 128 * 128; i += blockDim.x) {
        bufA[i] = (__bf16)latent[i];
        h0[i]   = (__bf16)0.0f;
    }
    for (int i = tid * 8; i < 384 * 128; i += 256 * 8) {   // weights -> LDS once
        *(v8bf*)&sWih[i] = *(const v8bf*)&Wih[i];
        *(v8bf*)&sWhh[i] = *(const v8bf*)&Whh[i];
    }
    __syncthreads();

    const int m0    = wave * 16;
    const int rowA  = m0 + (lane & 15);        // A-frag row
    const int khA   = (lane >> 4) * 8;         // A-frag K sub-offset
    const int kB    = (lane >> 4) * 8;         // B-frag K sub-offset
    const int colC  = lane & 15;               // C/D column
    const int rowC0 = m0 + ((lane >> 4) * 8);  // C/D rows: rowC0+v
    int lenC[8];
#pragma unroll
    for (int v = 0; v < 8; v++) lenC[v] = lengths[rowC0 + v];

    for (int t = 0; t < T_; t++) {
        const __bf16* hc = (t & 1) ? h1 : h0;
        __bf16*       hn = (t & 1) ? h0 : h1;
        const __bf16* xa = (t == 0) ? bufA : hc;

        for (int u0 = 0; u0 < 8; u0++) {
            const int u = u0 * 16;
            v8f ax3[3] = {}, ah3[3] = {};
#pragma unroll
            for (int kk = 0; kk < 4; kk++) {
                const int k0 = kk * 32;
                v16bf ax = mk16(*(const v8bf*)&xa[rowA * 128 + k0 + khA],
                                *(const v8bf*)&xa[rowA * 128 + k0 + 16 + khA]);
                v16bf ah = mk16(*(const v8bf*)&hc[rowA * 128 + k0 + khA],
                                *(const v8bf*)&hc[rowA * 128 + k0 + 16 + khA]);
#pragma unroll
                for (int g = 0; g < 3; g++) {
                    const int r = g * 128 + u + colC;
                    v16bf bx = mk16(*(const v8bf*)&sWih[r * 128 + k0 + kB],
                                    *(const v8bf*)&sWih[r * 128 + k0 + kB + 16]);
                    v16bf bh = mk16(*(const v8bf*)&sWhh[r * 128 + k0 + kB],
                                    *(const v8bf*)&sWhh[r * 128 + k0 + kB + 16]);
                    ax3[g] = __builtin_amdgcn_wmma_f32_16x16x32_bf16(
                        false, ax, false, bx, (short)0, ax3[g], false, false);
                    ah3[g] = __builtin_amdgcn_wmma_f32_16x16x32_bf16(
                        false, ah, false, bh, (short)0, ah3[g], false, false);
                }
            }
            const int j = u + colC;
            const float bxr = bih[j], bxz = bih[128 + j], bxn = bih[256 + j];
            const float bhr = bhh[j], bhz = bhh[128 + j], bhn = bhh[256 + j];
#pragma unroll
            for (int v = 0; v < 8; v++) {
                const int b = rowC0 + v;
                float r = sigf(ax3[0][v] + bxr + ah3[0][v] + bhr);
                float z = sigf(ax3[1][v] + bxz + ah3[1][v] + bhz);
                float n = tanhf(ax3[2][v] + bxn + r * (ah3[2][v] + bhn));
                float hp = (float)hc[b * 128 + j];
                float hv = (1.0f - z) * n + z * hp;
                hn[b * 128 + j] = (__bf16)hv;
                y0[((long)b * T_ + t) * IN_ + j] =
                    (t < lenC[v]) ? (__bf16)hv : (__bf16)0.0f;
            }
        }
        __syncthreads();
    }
}

// ---------------------------------------------------------------------------
// One biLSTM timestep, both directions.  grid = (50 unit-tiles, 2 dirs),
// block 256 = 8 waves; wave = batch-tile.  The shared B tile (4 gates x
// 16 rows x 32 k, 4KB) is DMA'd into LDS by the Tensor Data Mover, double
// buffered: TDM of chunk k+1 overlaps WMMA on chunk k.  A fragments come
// straight from global ([x_t | h_{t-1}], K = din+800).
// ---------------------------------------------------------------------------
__global__ __launch_bounds__(256) void lstm_step_kernel(
    int t, int din, const __bf16* __restrict__ xin /*[B][T][din]*/,
    const __bf16* __restrict__ W0, const __bf16* __restrict__ W1,
    const float* __restrict__ bs0, const float* __restrict__ bs1,
    const __bf16* __restrict__ hin /*[2][B][H]*/, __bf16* __restrict__ hout,
    float* __restrict__ cstate /*[2][B][H]*/, const int* __restrict__ lengths,
    __bf16* __restrict__ yout /*[B][T][2H]*/) {
    __shared__ __bf16 sB[2][4 * 16 * 32];   // [buf][gate][row][k]

    const int dir = blockIdx.y;
    const int n0  = blockIdx.x * 16;
    const int tid = threadIdx.x, lane = tid & 31, wave = tid >> 5;
    const int Kt  = din + H_;
    const int nk  = Kt / 32;
    const __bf16* W  = dir ? W1 : W0;
    const float*  bs = dir ? bs1 : bs0;
    const __bf16* hi = hin + (long)dir * B_ * H_;
    __bf16*       ho = hout + (long)dir * B_ * H_;
    float*        cb = cstate + (long)dir * B_ * H_;

    const int m0    = wave * 16;
    const int rowA  = m0 + (lane & 15);
    const int lenA  = lengths[rowA];
    const int sA    = dir ? ((t < lenA) ? lenA - 1 - t : t) : t;  // rev_idx
    const __bf16* xrow = xin + ((long)rowA * T_ + sA) * din;
    const __bf16* hrow = hi + (long)rowA * H_;
    const int khA   = (lane >> 4) * 8;
    const int kB    = (lane >> 4) * 8;
    const int colC  = lane & 15;
    const int rowC0 = m0 + ((lane >> 4) * 8);

#ifdef USE_TDM
    if (wave == 0) tdm_load_B(W, Kt, n0, 0, lds_off_of(&sB[0][0]));
#endif

    v8f acc[4] = {};
    for (int kk = 0; kk < nk; kk++) {
        const int k0 = kk * 32, bi = kk & 1;
#ifdef USE_TDM
        if (kk + 1 < nk) {
            if (wave == 0)
                tdm_load_B(W, Kt, n0, k0 + 32, lds_off_of(&sB[(kk + 1) & 1][0]));
            __builtin_amdgcn_s_wait_tensorcnt(1);   // chunk kk landed
        } else {
            __builtin_amdgcn_s_wait_tensorcnt(0);
        }
        __syncthreads();
#else
        {   // cooperative synchronous staging fallback
            const int g = tid >> 6, row = (tid >> 2) & 15, kp = (tid & 3) * 8;
            const __bf16* src = W + (long)(g * H_ + n0 + row) * Kt + k0 + kp;
            *(v8bf*)&sB[bi][(g * 16 + row) * 32 + kp] = *(const v8bf*)src;
            __syncthreads();
        }
#endif
        // A fragment shared by the 4 gate WMMAs
        const int ka = k0 + khA, kb2 = k0 + 16 + khA;
        const __bf16* s1 = (ka  < din) ? (xrow + ka)  : (hrow + (ka  - din));
        const __bf16* s2 = (kb2 < din) ? (xrow + kb2) : (hrow + (kb2 - din));
        v16bf a = mk16(*(const v8bf*)s1, *(const v8bf*)s2);
#pragma unroll
        for (int g = 0; g < 4; g++) {
            const __bf16* bp = &sB[bi][(g * 16 + colC) * 32 + kB];
            v16bf bf = mk16(*(const v8bf*)bp, *(const v8bf*)(bp + 16));
            acc[g] = __builtin_amdgcn_wmma_f32_16x16x32_bf16(
                false, a, false, bf, (short)0, acc[g], false, false);
        }
        __syncthreads();   // buffer bi may be re-staged next iteration
    }

    const int j = n0 + colC;
    const float bi_ = bs[j], bf_ = bs[H_ + j], bg = bs[2 * H_ + j], bo = bs[3 * H_ + j];
#pragma unroll
    for (int v = 0; v < 8; v++) {
        const int b2 = rowC0 + v;
        float i  = sigf(acc[0][v] + bi_);
        float f  = sigf(acc[1][v] + bf_);
        float g  = tanhf(acc[2][v] + bg);
        float o  = sigf(acc[3][v] + bo);
        float cO = cb[(long)b2 * H_ + j];
        float cN = f * cO + i * g;
        cb[(long)b2 * H_ + j] = cN;
        float h = o * tanhf(cN);
        ho[(long)b2 * H_ + j] = (__bf16)h;
        const int len2 = lengths[b2];
        const int s2   = dir ? ((t < len2) ? len2 - 1 - t : t) : t;
        yout[((long)b2 * T_ + s2) * (2 * H_) + dir * H_ + j] =
            (t < len2) ? (__bf16)h : (__bf16)0.0f;
    }
}

// ---------------------------------------------------------------------------
// AngleProject: atan2(x@sin_w^T + sin_b, x@cos_w^T + cos_b) * mask
// ---------------------------------------------------------------------------
__global__ __launch_bounds__(256) void angle_kernel(
    const __bf16* __restrict__ x /*[B][T][2H]*/, const float* __restrict__ sw,
    const float* __restrict__ sb, const float* __restrict__ cw,
    const float* __restrict__ cbv, const int* __restrict__ lengths,
    float* __restrict__ out /*[B][T][3]*/) {
    const int lane = threadIdx.x & 31, wave = threadIdx.x >> 5;
    const int pair = blockIdx.x * 8 + wave;  // flat (b,t)
    const int b = pair / T_, t = pair % T_;
    const __bf16* xr = x + (long)pair * (2 * H_);
    float s[3] = {0, 0, 0}, c[3] = {0, 0, 0};
    for (int k = lane; k < 2 * H_; k += 32) {
        float xv = (float)xr[k];
#pragma unroll
        for (int j = 0; j < 3; j++) {
            s[j] += xv * sw[j * 2 * H_ + k];
            c[j] += xv * cw[j * 2 * H_ + k];
        }
    }
#pragma unroll
    for (int j = 0; j < 3; j++) {
#pragma unroll
        for (int off = 16; off > 0; off >>= 1) {
            s[j] += __shfl_down(s[j], off, 32);
            c[j] += __shfl_down(c[j], off, 32);
        }
    }
    if (lane == 0) {
        float m = (t < lengths[b]) ? 1.0f : 0.0f;
#pragma unroll
        for (int j = 0; j < 3; j++)
            out[(long)pair * 3 + j] = atan2f(s[j] + sb[j], c[j] + cbv[j]) * m;
    }
}

// ---------------------------------------------------------------------------
extern "C" void kernel_launch(void* const* d_in, const int* in_sizes, int n_in,
                              void* d_out, int out_size, void* d_ws, size_t ws_size,
                              hipStream_t stream) {
    (void)in_sizes; (void)n_in; (void)out_size; (void)ws_size;
    const float* latent = (const float*)d_in[0];
    const float* gwih   = (const float*)d_in[1];
    const float* gwhh   = (const float*)d_in[2];
    const float* gbih   = (const float*)d_in[3];
    const float* gbhh   = (const float*)d_in[4];
    const float *Wih[4], *Whh[4], *bih[4], *bhh[4];
    for (int lp = 0; lp < 4; lp++) {  // lp = layer*2 + dir
        Wih[lp] = (const float*)d_in[5 + 4 * lp + 0];
        Whh[lp] = (const float*)d_in[5 + 4 * lp + 1];
        bih[lp] = (const float*)d_in[5 + 4 * lp + 2];
        bhh[lp] = (const float*)d_in[5 + 4 * lp + 3];
    }
    const float* sw  = (const float*)d_in[21];
    const float* sb  = (const float*)d_in[22];
    const float* cw  = (const float*)d_in[23];
    const float* cbv = (const float*)d_in[24];
    const int* lengths = (const int*)d_in[25];

    char* ws = (char*)d_ws;
    size_t off = 0;
    auto alloc = [&](size_t bytes) -> char* {
        char* p = ws + off;
        off += (bytes + 255) & ~(size_t)255;
        return p;
    };
    __bf16* gWih = (__bf16*)alloc((size_t)384 * 128 * 2);
    __bf16* gWhh = (__bf16*)alloc((size_t)384 * 128 * 2);
    __bf16* Wc[4];
    int Ktl[4];
    for (int lp = 0; lp < 4; lp++) {
        int din = (lp < 2) ? IN_ : 2 * H_;
        Ktl[lp] = din + H_;
        Wc[lp]  = (__bf16*)alloc((size_t)G4_ * Ktl[lp] * 2);
    }
    float* bsum[4];
    for (int lp = 0; lp < 4; lp++) bsum[lp] = (float*)alloc((size_t)G4_ * 4);
    __bf16* x0 = (__bf16*)alloc((size_t)B_ * T_ * IN_ * 2);
    __bf16* x1 = (__bf16*)alloc((size_t)B_ * T_ * 2 * H_ * 2);
    __bf16* x2 = (__bf16*)alloc((size_t)B_ * T_ * 2 * H_ * 2);
    __bf16* hpp[2];
    hpp[0] = (__bf16*)alloc((size_t)2 * B_ * H_ * 2);
    hpp[1] = (__bf16*)alloc((size_t)2 * B_ * H_ * 2);
    float* cst = (float*)alloc((size_t)2 * B_ * H_ * 4);

    // ---- pack weights to bf16 (recomputed every call: deterministic) ----
    cvt_f32_bf16<<<(384 * 128 + 255) / 256, 256, 0, stream>>>(gWih, gwih, 384 * 128);
    cvt_f32_bf16<<<(384 * 128 + 255) / 256, 256, 0, stream>>>(gWhh, gwhh, 384 * 128);
    for (int lp = 0; lp < 4; lp++) {
        int din = (lp < 2) ? IN_ : 2 * H_;
        pack_wcat<<<1024, 256, 0, stream>>>(Wc[lp], Wih[lp], Whh[lp], din);
        bias_sum<<<(G4_ + 255) / 256, 256, 0, stream>>>(bsum[lp], bih[lp], bhh[lp], G4_);
    }

    // ---- autoregressive GRU: single WG, 288KB dynamic LDS (weights + h) ----
    gru_kernel<<<1, 256, (3 * 16384 + 2 * 49152) * sizeof(__bf16), stream>>>(
        latent, gWih, gWhh, gbih, gbhh, lengths, x0);

    // ---- two biLSTM layers, one launch per timestep ----
    for (int layer = 0; layer < 2; layer++) {
        const int din = (layer == 0) ? IN_ : 2 * H_;
        const __bf16* xin = (layer == 0) ? x0 : x1;
        __bf16* yout      = (layer == 0) ? x1 : x2;
        (void)hipMemsetAsync(hpp[0], 0, (size_t)2 * B_ * H_ * 2, stream);
        (void)hipMemsetAsync(cst, 0, (size_t)2 * B_ * H_ * 4, stream);
        for (int t = 0; t < T_; t++) {
            lstm_step_kernel<<<dim3(H_ / 16, 2), 256, 0, stream>>>(
                t, din, xin, Wc[layer * 2], Wc[layer * 2 + 1], bsum[layer * 2],
                bsum[layer * 2 + 1], hpp[t & 1], hpp[(t + 1) & 1], cst, lengths, yout);
        }
    }

    // ---- angle projection ----
    angle_kernel<<<B_ * T_ / 8, 256, 0, stream>>>(x2, sw, sb, cw, cbv, lengths,
                                                  (float*)d_out);
}
